// RotationLocalGeoDynamicLayer_37048387895890
// MI455X (gfx1250) — matrix operand order repaired and checked by reading, not compiled, not run on hardware
//
#include <hip/hip_runtime.h>

// ---------------------------------------------------------------------------
// RotationLocalGeoDynamicLayer for MI455X (gfx1250, wave32, WMMA bf16)
//
// Woodbury restructure: transform_b = I + U C_b M_b^{-1} U^T with
//   U   = [L_1 R_1 ... L_8 R_8]  (384 x 128), batch independent
//   C_b = blockdiag([[0, c_k I8],[-c_k I8, 0]]),  c_k = tanh(rot)_k / norm_k
//   M_b = I_128 - 0.5 * (U^T U) C_b              (128 x 128, ~I, GJ-invert in LDS)
// rotated = lm + (lm U) (C_b M_b^{-1}) U^T  -> three bf16-WMMA GEMMs + final GEMM.
// ---------------------------------------------------------------------------

#define BB   64
#define NN   1025
#define DD   384
#define KK   8
#define RR   8
#define HH   128
#define CIN  1152
#define ROWS (BB * NN)   // 65600 = 1025 * 64 (multiple of 64)

typedef __attribute__((ext_vector_type(16))) __bf16 v16bf;
typedef __attribute__((ext_vector_type(8)))  float  v8f;

union Frag16 { v16bf v; uint4 u[2]; };
static_assert(sizeof(Frag16) == 32, "frag size");
union Pack4 { __bf16 h[4]; uint2 u; };
static_assert(sizeof(Pack4) == 8, "pack size");

// ---------------- stats: z = [cls, mean, var] ------------------------------
__global__ void k_stats(const float* __restrict__ x, float* __restrict__ z) {
    int b = blockIdx.x, d = threadIdx.x;                 // 384 threads
    long long base = (long long)b * NN * DD;
    float s = 0.f, sq = 0.f;
    for (int n = 0; n < NN; ++n) {
        float v = x[base + (long long)n * DD + d];
        s += v; sq += v * v;
    }
    float mean = s * (1.f / NN);
    float var  = sq * (1.f / NN) - mean * mean;
    z[b * CIN + d]           = x[base + d];
    z[b * CIN + DD + d]      = mean;
    z[b * CIN + 2 * DD + d]  = var;
}

// ---------------- Frobenius norms of skew bases via Gram identity ----------
__global__ void k_norm(const float* __restrict__ L, const float* __restrict__ R,
                       float* __restrict__ norms) {
    int k = blockIdx.x, t = threadIdx.x;                 // 64 threads
    int r = t >> 3, s = t & 7;
    const float* Lk = L + k * DD * RR;
    const float* Rk = R + k * DD * RR;
    float a = 0.f, g = 0.f, x1 = 0.f, x2 = 0.f;
    for (int d = 0; d < DD; ++d) {
        float lr = Lk[d * RR + r], ls = Lk[d * RR + s];
        float rr = Rk[d * RR + r], rs = Rk[d * RR + s];
        a  += lr * ls;   g  += rr * rs;
        x1 += lr * rs;   x2 += ls * rr;
    }
    __shared__ float red[64];
    red[t] = 2.f * (a * g - x1 * x2);
    __syncthreads();
    if (t == 0) {
        float sum = 0.f;
        for (int i = 0; i < 64; ++i) sum += red[i];
        norms[k] = fmaxf(sqrtf(fmaxf(sum, 0.f)), 1e-6f);
    }
}

// ---------------- controller MLP + heads -----------------------------------
__global__ void k_ctrl(const float* __restrict__ z,
                       const float* __restrict__ w1, const float* __restrict__ b1,
                       const float* __restrict__ w_rot, const float* __restrict__ b_rot,
                       const float* __restrict__ w_rg,  const float* __restrict__ b_rg,
                       const float* __restrict__ w_lg,  const float* __restrict__ b_lg,
                       const float* __restrict__ w_sc,  const float* __restrict__ b_sc,
                       const float* __restrict__ norms,
                       float* __restrict__ rc, float* __restrict__ rg,
                       float* __restrict__ lg, float* __restrict__ sc) {
    int b = blockIdx.x, t = threadIdx.x;                 // 128 threads
    __shared__ float zs[CIN];
    __shared__ float hs[HH];
    for (int i = t; i < CIN; i += HH) zs[i] = z[b * CIN + i];
    __syncthreads();
    float acc = b1[t];
    for (int i = 0; i < CIN; ++i) acc += zs[i] * w1[t * CIN + i];
    float u = 0.7978845608028654f * (acc + 0.044715f * acc * acc * acc);
    hs[t] = 0.5f * acc * (1.f + tanhf(u));               // gelu (tanh approx)
    __syncthreads();
    if (t < KK) {
        float ar = b_rot[t], as = b_sc[t];
        for (int i = 0; i < HH; ++i) { ar += hs[i] * w_rot[t * HH + i]; as += hs[i] * w_sc[t * HH + i]; }
        rc[b * KK + t] = tanhf(ar) / norms[t];           // fold 1/norm into coeff
        sc[b * KK + t] = tanhf(as);
    } else if (t == 8) {
        float a = b_rg[0];
        for (int i = 0; i < HH; ++i) a += hs[i] * w_rg[i];
        rg[b] = 1.f / (1.f + expf(-a));
    } else if (t == 9) {
        float a = b_lg[0];
        for (int i = 0; i < HH; ++i) a += hs[i] * w_lg[i];
        lg[b] = 1.f / (1.f + expf(-a));
    }
}

// ---------------- per-batch output scale -----------------------------------
__global__ void k_scale(const float* __restrict__ sc, const float* __restrict__ sb,
                        const float* __restrict__ ss, float* __restrict__ scale) {
    int b = blockIdx.x, o = threadIdx.x;                 // 384 threads
    float acc = 0.f;
    for (int k = 0; k < KK; ++k) acc += sc[b * KK + k] * sb[k * DD + o];
    scale[b * DD + o] = 1.f + ss[0] * tanhf(acc);
}

// ---------------- build U = [L_1 R_1 ... L_8 R_8] --------------------------
__global__ void k_buildU(const float* __restrict__ L, const float* __restrict__ R,
                         float* __restrict__ U) {
    int d = blockIdx.x, j = threadIdx.x;                 // 128 threads
    int k = j >> 4, s = j & 15;
    U[d * 128 + j] = (s < 8) ? L[k * DD * RR + d * RR + s]
                             : R[k * DD * RR + d * RR + (s - 8)];
}

// ---------------- G = U^T U -------------------------------------------------
__global__ void k_gram(const float* __restrict__ U, float* __restrict__ Gm) {
    int i = blockIdx.y * 16 + threadIdx.y;
    int j = blockIdx.x * 16 + threadIdx.x;
    float acc = 0.f;
    for (int d = 0; d < DD; ++d) acc += U[d * 128 + i] * U[d * 128 + j];
    Gm[i * 128 + j] = acc;
}

// ---------------- per-batch E = C M^{-1}, M = I - 0.5 G C (GJ in LDS) -------
__global__ __launch_bounds__(256) void k_minv(const float* __restrict__ Gm,
                                              const float* __restrict__ rc,
                                              float* __restrict__ E) {
    int b = blockIdx.x, t = threadIdx.x;                 // 256 threads
    __shared__ float A[128][128];                        // 64 KB LDS
    // Build M: column 16k+s picks -c*G[:,16k+8+s]; column 16k+8+s picks +c*G[:,16k+s]
    for (int idx = t; idx < 128 * 128; idx += 256) {
        int i = idx >> 7, j = idx & 127;
        int k = j >> 4, s = j & 15;
        float c = rc[b * KK + k];
        float g = (s < 8) ? Gm[i * 128 + (j + 8)] : Gm[i * 128 + (j - 8)];
        float val = (s < 8) ? 0.5f * c * g : -0.5f * c * g;
        if (i == j) val += 1.f;
        A[i][j] = val;
    }
    __syncthreads();
    // In-place Gauss-Jordan inversion (M ~ I, diagonally dominant: no pivoting)
    for (int p = 0; p < 128; ++p) {
        float inv = 1.f / A[p][p];
        __syncthreads();
        if (t < 128 && t != p) A[p][t] *= inv;
        __syncthreads();
        for (int idx = t; idx < 128 * 128; idx += 256) {
            int i = idx >> 7, j = idx & 127;
            if (i != p && j != p) A[i][j] -= A[i][p] * A[p][j];
        }
        __syncthreads();
        if (t < 128) {
            if (t != p) A[t][p] = -A[t][p] * inv; else A[p][p] = inv;
        }
        __syncthreads();
    }
    // E = C * Minv: row 16k+s = +c*Minv[16k+8+s,:], row 16k+8+s = -c*Minv[16k+s,:]
    for (int idx = t; idx < 128 * 128; idx += 256) {
        int i = idx >> 7, j = idx & 127;
        int k = i >> 4, s = i & 15;
        float c = rc[b * KK + k];
        float val = (s < 8) ? c * A[i + 8][j] : -c * A[i - 8][j];
        E[(long long)b * 16384 + i * 128 + j] = val;
    }
}

// ---------------- lm: cls passthrough + depthwise 3x3 conv mix -------------
__global__ void k_lm(const float* __restrict__ x, const float* __restrict__ dwk,
                     const float* __restrict__ lg, float* __restrict__ lm) {
    int n = blockIdx.x, b = blockIdx.y, d = threadIdx.x; // 384 threads
    long long base = (long long)b * NN * DD;
    if (n == 0) { lm[base + d] = x[base + d]; return; }
    int p = n - 1, gy = p >> 5, gx = p & 31;
    float acc = 0.f;
#pragma unroll
    for (int dy = -1; dy <= 1; ++dy) {
        int yy = gy + dy;
        if (yy < 0 || yy > 31) continue;
#pragma unroll
        for (int dx = -1; dx <= 1; ++dx) {
            int xx = gx + dx;
            if (xx < 0 || xx > 31) continue;
            float v = x[base + (long long)(1 + yy * 32 + xx) * DD + d];
            acc += v * dwk[d * 9 + (dy + 1) * 3 + (dx + 1)];
        }
    }
    float c = x[base + (long long)n * DD + d];
    lm[base + (long long)n * DD + d] = c + lg[b] * (acc - c);
}

// ---------------- bf16 WMMA tiled GEMM -------------------------------------
// Block tile 64(M) x 128(N), K-chunk 64. 8 waves (2x4); each wave owns a 2x2
// grid of 16x16 accumulators (M-subtiles 16 apart, N-subtiles 64 apart) and
// issues 8 WMMAs per K-chunk. Staging is two-phase: all global b128 loads
// first (kept in flight together), then cvt+LDS stores.
// epi 0: C = acc          epi 1: C += rg[row/NN]*acc (mix into lm)
// epi 2: C = (acc + bias[col]) * scale[row/NN][col]
__global__ __launch_bounds__(256) void k_gemm(
    const float* __restrict__ Ag, int lda, long long aStr,
    const float* __restrict__ Bg, int ldb, long long bStr, int bT,
    float* __restrict__ Cg, int ldc, long long cStr,
    int M, int Ktot, int epi,
    const float* __restrict__ aux1, const float* __restrict__ aux2) {
    Ag += blockIdx.z * aStr;  Bg += blockIdx.z * bStr;  Cg += blockIdx.z * cStr;
    const int tid  = threadIdx.x;
    const int wave = tid >> 5, lane = tid & 31;
    const int wm = wave >> 2, wn = wave & 3;             // 2 x 4 waves
    const int m16 = lane & 15, half = lane >> 4;
    const int rowBase = blockIdx.y * 64;
    const int colBase = blockIdx.x * 128;

    __shared__ __bf16 As[64][64];                        // [m][k]   8 KB
    __shared__ __bf16 Bs[128][64];                       // [n][k]  16 KB (transposed)

    v8f acc[2][2];
#pragma unroll
    for (int i = 0; i < 2; ++i)
#pragma unroll
        for (int j = 0; j < 2; ++j)
            acc[i][j] = (v8f){0.f, 0.f, 0.f, 0.f, 0.f, 0.f, 0.f, 0.f};

    for (int k0 = 0; k0 < Ktot; k0 += 64) {
        // ---- phase 1: issue ALL global tile loads (independent, stay in flight)
        float4 ra[4];
#pragma unroll
        for (int i = 0; i < 4; ++i) {
            int q = tid + i * 256;                       // A: 64x64 -> 1024 float4
            int r = q >> 4;
            int c = (q & 15) << 2;
            int row  = rowBase + r;
            int rowc = row < M ? row : M - 1;            // branchless clamp
            ra[i] = *(const float4*)&Ag[(long long)rowc * lda + k0 + c];
        }
        float4 rb[8];
        if (bT) {        // B is (N x K) row-major: float4 along k
#pragma unroll
            for (int i = 0; i < 8; ++i) {
                int q = tid + i * 256;                   // B: 128x64 -> 2048 float4
                int n = q >> 4;
                int c = (q & 15) << 2;
                rb[i] = *(const float4*)&Bg[(long long)(colBase + n) * ldb + k0 + c];
            }
        } else {         // B is (K x N) row-major: float4 along n
#pragma unroll
            for (int i = 0; i < 8; ++i) {
                int q = tid + i * 256;
                int k = q >> 5;                          // 32 float4 per k-row
                int c = (q & 31) << 2;
                rb[i] = *(const float4*)&Bg[(long long)(k0 + k) * ldb + colBase + c];
            }
        }
        // ---- phase 2: convert + store to LDS ----
#pragma unroll
        for (int i = 0; i < 4; ++i) {
            int q = tid + i * 256;
            int r = q >> 4;
            int c = (q & 15) << 2;
            bool ok = (rowBase + r) < M;
            Pack4 p;
            p.h[0] = (__bf16)(ok ? ra[i].x : 0.f);
            p.h[1] = (__bf16)(ok ? ra[i].y : 0.f);
            p.h[2] = (__bf16)(ok ? ra[i].z : 0.f);
            p.h[3] = (__bf16)(ok ? ra[i].w : 0.f);
            *(uint2*)&As[r][c] = p.u;
        }
        if (bT) {
#pragma unroll
            for (int i = 0; i < 8; ++i) {
                int q = tid + i * 256;
                int n = q >> 4;
                int c = (q & 15) << 2;
                Pack4 p;
                p.h[0] = (__bf16)rb[i].x; p.h[1] = (__bf16)rb[i].y;
                p.h[2] = (__bf16)rb[i].z; p.h[3] = (__bf16)rb[i].w;
                *(uint2*)&Bs[n][c] = p.u;
            }
        } else {
#pragma unroll
            for (int i = 0; i < 8; ++i) {
                int q = tid + i * 256;
                int k = q >> 5;
                int c = (q & 31) << 2;
                Bs[c + 0][k] = (__bf16)rb[i].x;
                Bs[c + 1][k] = (__bf16)rb[i].y;
                Bs[c + 2][k] = (__bf16)rb[i].z;
                Bs[c + 3][k] = (__bf16)rb[i].w;
            }
        }
        __syncthreads();

#pragma unroll
        for (int kb = 0; kb < 64; kb += 32) {
            // A frags: lane(m,h): K = kb + 8h..+7 then kb+16+8h..+7 (ISA layout)
            Frag16 a0, a1;
            a0.u[0] = *(const uint4*)&As[wm * 32 + m16][kb + half * 8];
            a0.u[1] = *(const uint4*)&As[wm * 32 + m16][kb + 16 + half * 8];
            a1.u[0] = *(const uint4*)&As[wm * 32 + 16 + m16][kb + half * 8];
            a1.u[1] = *(const uint4*)&As[wm * 32 + 16 + m16][kb + 16 + half * 8];
            // B frags: lane(n,h): K = kb + 8h..+7 then kb+16+8h..+7; n-subtiles 64 apart
            Frag16 b0, b1;
            b0.u[0] = *(const uint4*)&Bs[wn * 16 + m16][kb + half * 8];
            b0.u[1] = *(const uint4*)&Bs[wn * 16 + m16][kb + 16 + half * 8];
            b1.u[0] = *(const uint4*)&Bs[64 + wn * 16 + m16][kb + half * 8];
            b1.u[1] = *(const uint4*)&Bs[64 + wn * 16 + m16][kb + 16 + half * 8];
            acc[0][0] = __builtin_amdgcn_wmma_f32_16x16x32_bf16(false, a0.v, false, b0.v,
                                                                (short)0, acc[0][0], false, false);
            acc[1][0] = __builtin_amdgcn_wmma_f32_16x16x32_bf16(false, a1.v, false, b0.v,
                                                                (short)0, acc[1][0], false, false);
            acc[0][1] = __builtin_amdgcn_wmma_f32_16x16x32_bf16(false, a0.v, false, b1.v,
                                                                (short)0, acc[0][1], false, false);
            acc[1][1] = __builtin_amdgcn_wmma_f32_16x16x32_bf16(false, a1.v, false, b1.v,
                                                                (short)0, acc[1][1], false, false);
        }
        __syncthreads();
    }

#pragma unroll
    for (int n2 = 0; n2 < 2; ++n2) {
        const int col = colBase + n2 * 64 + wn * 16 + m16;
#pragma unroll
        for (int m2 = 0; m2 < 2; ++m2) {
#pragma unroll
            for (int v = 0; v < 8; ++v) {
                int row = rowBase + wm * 32 + m2 * 16 + half * 8 + v;  // C/D: vgpr v -> M = v + 8h
                if (row < M) {
                    long long o = (long long)row * ldc + col;
                    float val = acc[m2][n2][v];
                    if (epi == 0)      Cg[o] = val;
                    else if (epi == 1) Cg[o] = Cg[o] + aux1[row / NN] * val;
                    else               Cg[o] = (val + aux1[col]) * aux2[(long long)(row / NN) * ldc + col];
                }
            }
        }
    }
}

// ---------------------------------------------------------------------------
extern "C" void kernel_launch(void* const* d_in, const int* in_sizes, int n_in,
                              void* d_out, int out_size, void* d_ws, size_t ws_size,
                              hipStream_t stream) {
    const float* x      = (const float*)d_in[0];
    const float* W      = (const float*)d_in[1];
    const float* bias   = (const float*)d_in[2];
    const float* Lb     = (const float*)d_in[3];
    const float* Rb     = (const float*)d_in[4];
    const float* sbas   = (const float*)d_in[5];
    const float* ss     = (const float*)d_in[6];
    const float* dwk    = (const float*)d_in[7];
    const float* w1     = (const float*)d_in[8];
    const float* b1     = (const float*)d_in[9];
    const float* w_rot  = (const float*)d_in[10];
    const float* b_rot  = (const float*)d_in[11];
    const float* w_rg   = (const float*)d_in[12];
    const float* b_rg   = (const float*)d_in[13];
    const float* w_lg   = (const float*)d_in[14];
    const float* b_lg   = (const float*)d_in[15];
    const float* w_sc   = (const float*)d_in[16];
    const float* b_sc   = (const float*)d_in[17];
    float* out = (float*)d_out;

    float* w = (float*)d_ws;
    float* zbuf  = w;  w += BB * CIN;            // 73728
    float* norms = w;  w += 64;
    float* rc    = w;  w += BB * KK;
    float* rgb   = w;  w += BB;
    float* lgb   = w;  w += BB;
    float* scb   = w;  w += BB * KK;
    float* scale = w;  w += BB * DD;             // 24576
    float* U     = w;  w += DD * 128;            // 49152
    float* Gm    = w;  w += 128 * 128;           // 16384
    float* E     = w;  w += (long long)BB * 128 * 128;     // 1.05M
    float* lm    = w;  w += (long long)ROWS * DD;          // 25.2M
    float* Q     = w;  w += (long long)ROWS * 128;         // 8.4M
    float* Q2    = w;  w += (long long)ROWS * 128;         // 8.4M

    // controller path
    k_stats<<<BB, DD, 0, stream>>>(x, zbuf);
    k_norm<<<KK, 64, 0, stream>>>(Lb, Rb, norms);
    k_ctrl<<<BB, HH, 0, stream>>>(zbuf, w1, b1, w_rot, b_rot, w_rg, b_rg,
                                  w_lg, b_lg, w_sc, b_sc, norms,
                                  rc, rgb, lgb, scb);
    k_scale<<<BB, DD, 0, stream>>>(scb, sbas, ss, scale);

    // low-rank rotation machinery
    k_buildU<<<DD, 128, 0, stream>>>(Lb, Rb, U);
    k_gram<<<dim3(8, 8), dim3(16, 16), 0, stream>>>(U, Gm);
    k_minv<<<BB, 256, 0, stream>>>(Gm, rc, E);

    // local (depthwise conv) mix -> lm
    k_lm<<<dim3(NN, BB), DD, 0, stream>>>(x, dwk, lgb, lm);

    // GEMM1: Q = lm @ U            (65600x384 @ 384x128)
    k_gemm<<<dim3(128 / 128, ROWS / 64, 1), 256, 0, stream>>>(
        lm, DD, 0, U, 128, 0, 0, Q, 128, 0, ROWS, DD, 0, nullptr, nullptr);
    // GEMM2: Q2_b = Q_b @ E_b      (per batch, 1025x128 @ 128x128)
    k_gemm<<<dim3(128 / 128, (NN + 63) / 64, BB), 256, 0, stream>>>(
        Q, 128, (long long)NN * 128, E, 128, 128 * 128, 0,
        Q2, 128, (long long)NN * 128, NN, 128, 0, nullptr, nullptr);
    // GEMM3: lm += rg * (Q2 @ U^T) (65600x128 @ 128x384, mix epilogue)
    k_gemm<<<dim3(DD / 128, ROWS / 64, 1), 256, 0, stream>>>(
        Q2, 128, 0, U, 128, 0, 1, lm, DD, 0, ROWS, 128, 1, rgb, nullptr);
    // GEMM4: out = (mixed @ W^T + bias) * scale
    k_gemm<<<dim3(DD / 128, ROWS / 64, 1), 256, 0, stream>>>(
        lm, DD, 0, W, DD, 0, 1, out, DD, 0, ROWS, DD, 2, bias, scale);

    (void)in_sizes; (void)n_in; (void)out_size; (void)ws_size;
}
